// CrossAttention_RoPE_26929444946151
// MI455X (gfx1250) — compile-verified, matrix operand order checked by reading, not run on hardware
//
#include <hip/hip_runtime.h>
#include <hip/hip_bf16.h>
#include <math.h>

// ---------------------------------------------------------------------------
// CDNA5 (gfx1250) cross-attention + RoPE, bf16 WMMA pipeline.
// Round 4: guard-free async fill path for interior tiles (raggedness decided
// once per block); async global->LDS double buffering; WMMA everywhere.
// ---------------------------------------------------------------------------

typedef __bf16 bf16_t;
typedef __attribute__((ext_vector_type(16))) __bf16 v16bf;
typedef __attribute__((ext_vector_type(8)))  float  v8f;

union FragB16 {
    uint4 u[2];   // two 16B chunks (8 bf16 each)
    v16bf v;      // WMMA operand
};
static_assert(sizeof(FragB16) == 32, "frag size");

#define BATCH   8
#define SEQ_L   1704
#define SEQ_LK  144
#define LKP     160      // Lk padded to multiple of 32 for WMMA K-dim
#define CDIM    1024
#define HEADS   16
#define HDIM    64
#define MAX_SMUL 4.6051701859880914f  // ln(100)

// --- async global->LDS (CDNA5 GLOBAL_LOAD_ASYNC_TO_LDS_B128) ---------------
#if defined(__has_builtin)
#if __has_builtin(__builtin_amdgcn_global_load_async_to_lds_b128) && \
    __has_builtin(__builtin_amdgcn_s_wait_asynccnt)
#define HAS_ASYNC_LDS 1
#endif
#endif
#ifndef HAS_ASYNC_LDS
#define HAS_ASYNC_LDS 0
#endif

__device__ __forceinline__ void async_copy16(const bf16_t* g, bf16_t* l) {
#if HAS_ASYNC_LDS
    typedef __attribute__((__vector_size__(4 * sizeof(int)))) int v4i_t;
    typedef __attribute__((address_space(1))) v4i_t as1_v4i;
    typedef __attribute__((address_space(3))) v4i_t as3_v4i;
    __builtin_amdgcn_global_load_async_to_lds_b128(
        (as1_v4i*)g, (as3_v4i*)l, 0, 0);
#else
    *reinterpret_cast<uint4*>(l) = *reinterpret_cast<const uint4*>(g);
#endif
}

__device__ __forceinline__ void wait_async_lds() {
#if HAS_ASYNC_LDS
    __builtin_amdgcn_s_wait_asynccnt(0);
#endif
}

// ---------------------------------------------------------------------------
// fp32 -> bf16 elementwise conversion
// ---------------------------------------------------------------------------
__global__ void cvt_f32_bf16(const float* __restrict__ in,
                             bf16_t* __restrict__ out, int n) {
    int i = blockIdx.x * blockDim.x + threadIdx.x;
    int stride = gridDim.x * blockDim.x;
    for (; i < n; i += stride) out[i] = (bf16_t)in[i];
}

// ---------------------------------------------------------------------------
// Cooperative tile copy: ROWS x 64 bf16 into LDS (pitch 72).
// GUARD=false: unpredicated async fill (interior tiles / weight tiles).
// GUARD=true : row-guarded; OOB rows zero-filled with ds stores (disjoint
//              addresses from the async writes; synced at the stage barrier).
// ---------------------------------------------------------------------------
template <int ROWS, bool GUARD>
__device__ __forceinline__ void load_tile(const bf16_t* __restrict__ g,
                                          int gRowBase, int M, int K, int k0,
                                          bf16_t* lds, int tid) {
    constexpr int CHUNKS = ROWS * 8;   // 16B chunks (8 bf16 each)
    constexpr int ITER = CHUNKS / 256;
#pragma unroll
    for (int it = 0; it < ITER; ++it) {
        int c   = tid + it * 256;
        int row = c >> 3;
        int col = (c & 7) * 8;
        int gr  = gRowBase + row;
        bf16_t* dst = lds + row * 72 + col;
        if (!GUARD || gr < M) {
            async_copy16(g + (size_t)gr * K + k0 + col, dst);
        } else {
            *reinterpret_cast<uint4*>(dst) = make_uint4(0u, 0u, 0u, 0u);
        }
    }
}

// ---------------------------------------------------------------------------
// Tiled bf16 WMMA GEMM:  C[M,N](f32) = A[M,K] * B[N,K]^T  (+ optional bias[N])
// A, B row-major bf16.  K % 64 == 0, N % 64 == 0, M guarded (ragged edge).
// Block = 256 threads = 8 waves (4x2); tile 128x64x64, double-buffered LDS,
// async global->LDS overlap; wave sub-tile 32x32 (2x2 WMMA tiles).
// ---------------------------------------------------------------------------
__global__ __launch_bounds__(256) void gemm_bf16_nt(
    const bf16_t* __restrict__ A, const bf16_t* __restrict__ B,
    const float* __restrict__ bias, float* __restrict__ C,
    int M, int N, int K)
{
    __shared__ __align__(16) bf16_t As[2][128 * 72];  // 2 x 18 KB
    __shared__ __align__(16) bf16_t Bs[2][64 * 72];   // 2 x  9 KB

    const int tid  = threadIdx.x;
    const int lane = tid & 31;
    const int wave = tid >> 5;
    const int mw   = wave >> 1;      // 0..3 -> 32-row slice
    const int nw   = wave & 1;       // 0..1 -> 32-col slice
    const int mBase = blockIdx.x * 128;
    const int nBase = blockIdx.y * 64;
    const int lrow  = lane & 15;
    const int khalf = (lane >> 4) * 8;
    const bool ragged = (mBase + 128 > M);   // uniform per block

    v8f acc[2][2];
#pragma unroll
    for (int i = 0; i < 2; ++i)
#pragma unroll
        for (int j = 0; j < 2; ++j)
#pragma unroll
            for (int r = 0; r < 8; ++r) acc[i][j][r] = 0.f;

    // prologue: stage 0
    if (ragged) load_tile<128, true >(A, mBase, M, K, 0, As[0], tid);
    else        load_tile<128, false>(A, mBase, M, K, 0, As[0], tid);
    load_tile<64, false>(B, nBase, N, K, 0, Bs[0], tid);
    wait_async_lds();
    __syncthreads();

    const int nk = K >> 6;
    for (int kt = 0; kt < nk; ++kt) {
        const int cur = kt & 1;
        const int nxt = cur ^ 1;
        // fill next stage while computing current one
        if (kt + 1 < nk) {
            const int k1 = (kt + 1) * 64;
            if (ragged) load_tile<128, true >(A, mBase, M, K, k1, As[nxt], tid);
            else        load_tile<128, false>(A, mBase, M, K, k1, As[nxt], tid);
            load_tile<64, false>(B, nBase, N, K, k1, Bs[nxt], tid);
        }
        if (kt + 2 < nk) {
            int pr = mBase + (tid >> 1);
            if (pr < M) __builtin_prefetch(A + (size_t)pr * K + (kt + 2) * 64, 0, 3);
        }

#pragma unroll
        for (int ks = 0; ks < 64; ks += 32) {
            FragB16 af[2], bfr[2];
#pragma unroll
            for (int i = 0; i < 2; ++i) {
                const bf16_t* p = &As[cur][(mw * 32 + i * 16 + lrow) * 72 + ks + khalf];
                af[i].u[0] = *reinterpret_cast<const uint4*>(p);
                af[i].u[1] = *reinterpret_cast<const uint4*>(p + 16);
            }
#pragma unroll
            for (int j = 0; j < 2; ++j) {
                const bf16_t* p = &Bs[cur][(nw * 32 + j * 16 + lrow) * 72 + ks + khalf];
                bfr[j].u[0] = *reinterpret_cast<const uint4*>(p);
                bfr[j].u[1] = *reinterpret_cast<const uint4*>(p + 16);
            }
#pragma unroll
            for (int i = 0; i < 2; ++i)
#pragma unroll
                for (int j = 0; j < 2; ++j)
                    acc[i][j] = __builtin_amdgcn_wmma_f32_16x16x32_bf16(
                        false, af[i].v, false, bfr[j].v,
                        (short)0, acc[i][j], false, false);
        }

        wait_async_lds();
        __syncthreads();
    }

    // C layout: VGPR r -> row r (lanes 0-15) / row r+8 (lanes 16-31), col = lane&15
#pragma unroll
    for (int i = 0; i < 2; ++i) {
        int rbase = mBase + mw * 32 + i * 16 + (lane >> 4) * 8;
#pragma unroll
        for (int j = 0; j < 2; ++j) {
            int col = nBase + nw * 32 + j * 16 + lrow;
            float bv = bias ? bias[col] : 0.f;
#pragma unroll
            for (int r = 0; r < 8; ++r) {
                int row = rbase + r;
                if (row < M) C[(size_t)row * N + col] = acc[i][j][r] + bv;
            }
        }
    }
}

// ---------------------------------------------------------------------------
// Q epilogue: bias + L2-normalize over D=64 + per-head scale + RoPE -> bf16
// One wave per (b,l,h); lane handles the (2*lane, 2*lane+1) RoPE pair.
// Output layout Qb[B,H,L,64].
// ---------------------------------------------------------------------------
__global__ __launch_bounds__(256) void q_epilogue(
    const float* __restrict__ Qf, const float* __restrict__ q_bias,
    const float* __restrict__ freqs, const float* __restrict__ scale_mul,
    bf16_t* __restrict__ Qb)
{
    const int lane = threadIdx.x & 31;
    const int wave = threadIdx.x >> 5;
    const int R = BATCH * SEQ_L * HEADS;
    int row = blockIdx.x * 8 + wave;
    if (row >= R) return;
    int h = row % HEADS;
    int l = (row / HEADS) % SEQ_L;
    int b = row / (HEADS * SEQ_L);

    size_t src = ((size_t)(b * SEQ_L + l)) * CDIM + h * HDIM + 2 * lane;
    float a  = Qf[src]     + q_bias[h * HDIM + 2 * lane];
    float bb = Qf[src + 1] + q_bias[h * HDIM + 2 * lane + 1];

    float s = a * a + bb * bb;
#pragma unroll
    for (int m = 16; m; m >>= 1) s += __shfl_xor(s, m, 32);
    float smul = __expf(fminf(scale_mul[h], MAX_SMUL));
    float rn = smul / fmaxf(sqrtf(s), 1e-12f);

    float c = freqs[(size_t)(l * 32 + lane) * 2];
    float d = freqs[(size_t)(l * 32 + lane) * 2 + 1];
    float ra = (a * c - bb * d) * rn;
    float rb = (a * d + bb * c) * rn;

    size_t dst = (((size_t)(b * HEADS + h)) * SEQ_L + l) * HDIM + 2 * lane;
    Qb[dst]     = (bf16_t)ra;
    Qb[dst + 1] = (bf16_t)rb;
}

// ---------------------------------------------------------------------------
// KV epilogue: K -> L2-normalize + cross-RoPE (12x12 grid recomputed on the
// fly) -> Kb[B,H,Lk,64]; V -> +v_bias, transposed -> Vt[B,H,64,LKP] (padded).
// ---------------------------------------------------------------------------
__global__ __launch_bounds__(256) void kv_epilogue(
    const float* __restrict__ KVf, const float* __restrict__ v_bias,
    bf16_t* __restrict__ Kb, bf16_t* __restrict__ Vt)
{
    const int lane = threadIdx.x & 31;
    const int wave = threadIdx.x >> 5;
    const int R = BATCH * SEQ_LK * HEADS;
    int row = blockIdx.x * 8 + wave;
    if (row >= R) return;
    int h  = row % HEADS;
    int lk = (row / HEADS) % SEQ_LK;
    int b  = row / (HEADS * SEQ_LK);

    // K part (no bias)
    size_t ksrc = ((size_t)(b * SEQ_LK + lk)) * (2 * CDIM) + h * HDIM + 2 * lane;
    float a  = KVf[ksrc];
    float bb = KVf[ksrc + 1];
    float s = a * a + bb * bb;
#pragma unroll
    for (int m = 16; m; m >>= 1) s += __shfl_xor(s, m, 32);
    float rn = 1.f / fmaxf(sqrtf(s), 1e-12f);

    // cross freqs: pos on 12x12 grid; tx = (pos/12)/12*32, ty = (pos%12)/12*32
    float t = (lane < 16) ? (float)(lk / 12) : (float)(lk % 12);
    t *= (32.f / 12.f);
    float fr  = __powf(10000.f, -(float)(lane & 15) * (1.f / 16.f));
    float ang = t * fr;
    float sn, cs;
    __sincosf(ang, &sn, &cs);
    float ra = (a * cs - bb * sn) * rn;
    float rb = (a * sn + bb * cs) * rn;

    size_t kdst = (((size_t)(b * HEADS + h)) * SEQ_LK + lk) * HDIM + 2 * lane;
    Kb[kdst]     = (bf16_t)ra;
    Kb[kdst + 1] = (bf16_t)rb;

    // V part: bias add + transpose into [B,H,64,LKP]
    size_t vsrc = ksrc + CDIM;
    float va = KVf[vsrc]     + v_bias[h * HDIM + 2 * lane];
    float vb = KVf[vsrc + 1] + v_bias[h * HDIM + 2 * lane + 1];
    size_t vdst = ((((size_t)(b * HEADS + h)) * HDIM) + 2 * lane) * LKP + lk;
    Vt[vdst]       = (bf16_t)va;
    Vt[vdst + LKP] = (bf16_t)vb;
}

// ---------------------------------------------------------------------------
// Fused attention: per wave one 16-row L tile for one (b,h).
// logits = Q*K^T (9 WMMA N-tiles over Lk=144) + bias; softmax in C-register
// layout (xor-shuffle row reductions); P staged to LDS bf16; O = P*V (K=160).
// ---------------------------------------------------------------------------
__global__ __launch_bounds__(256) void attn_kernel(
    const bf16_t* __restrict__ Qb, const bf16_t* __restrict__ Kb,
    const bf16_t* __restrict__ Vt, const float* __restrict__ attn_bias,
    bf16_t* __restrict__ Ob)
{
    __shared__ __align__(16) bf16_t Ps[8 * 16 * LKP];   // 40 KB

    const int tid  = threadIdx.x;
    const int lane = tid & 31;
    const int wave = tid >> 5;
    const int h = blockIdx.y;
    const int b = blockIdx.z;
    const int m0 = blockIdx.x * 128 + wave * 16;
    const int lrow  = lane & 15;
    const int khalf = (lane >> 4) * 8;
    const int rrow0 = m0 + (lane >> 4) * 8;

    // --- Q A-fragments (D=64 -> 2 K-steps) ---
    FragB16 qa[2];
    {
        int gr = m0 + lrow;
        const bf16_t* qp = Qb + (((size_t)(b * HEADS + h)) * SEQ_L + gr) * HDIM;
#pragma unroll
        for (int ks = 0; ks < 2; ++ks) {
            if (gr < SEQ_L) {
                qa[ks].u[0] = *reinterpret_cast<const uint4*>(qp + ks * 32 + khalf);
                qa[ks].u[1] = *reinterpret_cast<const uint4*>(qp + ks * 32 + khalf + 16);
            } else {
                qa[ks].u[0] = make_uint4(0u, 0u, 0u, 0u);
                qa[ks].u[1] = make_uint4(0u, 0u, 0u, 0u);
            }
        }
    }

    // --- logits = Q * K^T : 9 N-tiles of 16 ---
    v8f acc[9];
#pragma unroll
    for (int t = 0; t < 9; ++t)
#pragma unroll
        for (int r = 0; r < 8; ++r) acc[t][r] = 0.f;

    const bf16_t* kbp = Kb + ((size_t)(b * HEADS + h)) * SEQ_LK * HDIM;
#pragma unroll
    for (int t = 0; t < 9; ++t) {
#pragma unroll
        for (int ks = 0; ks < 2; ++ks) {
            FragB16 kf;
            const bf16_t* kp = kbp + (t * 16 + lrow) * HDIM + ks * 32 + khalf;
            kf.u[0] = *reinterpret_cast<const uint4*>(kp);
            kf.u[1] = *reinterpret_cast<const uint4*>(kp + 16);
            acc[t] = __builtin_amdgcn_wmma_f32_16x16x32_bf16(
                false, qa[ks].v, false, kf.v, (short)0, acc[t], false, false);
        }
    }

    // --- + attn_bias, row max ---
    float mx[8];
#pragma unroll
    for (int r = 0; r < 8; ++r) mx[r] = -1e30f;
#pragma unroll
    for (int t = 0; t < 9; ++t) {
        int col = t * 16 + lrow;
#pragma unroll
        for (int r = 0; r < 8; ++r) {
            int gr = rrow0 + r;
            float bv = (gr < SEQ_L) ? attn_bias[(size_t)gr * SEQ_LK + col] : 0.f;
            float v = acc[t][r] + bv;
            acc[t][r] = v;
            mx[r] = fmaxf(mx[r], v);
        }
    }
#pragma unroll
    for (int r = 0; r < 8; ++r)
#pragma unroll
        for (int m = 8; m; m >>= 1) mx[r] = fmaxf(mx[r], __shfl_xor(mx[r], m, 32));

    // --- exp + row sum ---
    float sum[8];
#pragma unroll
    for (int r = 0; r < 8; ++r) sum[r] = 0.f;
#pragma unroll
    for (int t = 0; t < 9; ++t)
#pragma unroll
        for (int r = 0; r < 8; ++r) {
            float p = __expf(acc[t][r] - mx[r]);
            acc[t][r] = p;
            sum[r] += p;
        }
#pragma unroll
    for (int r = 0; r < 8; ++r)
#pragma unroll
        for (int m = 8; m; m >>= 1) sum[r] += __shfl_xor(sum[r], m, 32);
    float inv[8];
#pragma unroll
    for (int r = 0; r < 8; ++r) inv[r] = 1.f / sum[r];

    // --- stage P (bf16) to LDS, zero-pad cols 144..159 ---
    bf16_t* ps = &Ps[wave * 16 * LKP];
#pragma unroll
    for (int t = 0; t < 9; ++t) {
        int col = t * 16 + lrow;
#pragma unroll
        for (int r = 0; r < 8; ++r) {
            int row = (lane >> 4) * 8 + r;
            ps[row * LKP + col] = (bf16_t)(acc[t][r] * inv[r]);
        }
    }
    *reinterpret_cast<uint4*>(&ps[lrow * LKP + SEQ_LK + khalf]) =
        make_uint4(0u, 0u, 0u, 0u);
    __syncthreads();

    // --- O = P * V : N=64 (4 tiles), K=160 (5 steps) ---
    v8f oacc[4];
#pragma unroll
    for (int j = 0; j < 4; ++j)
#pragma unroll
        for (int r = 0; r < 8; ++r) oacc[j][r] = 0.f;

    const bf16_t* vtp = Vt + ((size_t)(b * HEADS + h)) * HDIM * LKP;
#pragma unroll
    for (int ks = 0; ks < 5; ++ks) {
        FragB16 pa;
        const bf16_t* pp = &ps[lrow * LKP + ks * 32 + khalf];
        pa.u[0] = *reinterpret_cast<const uint4*>(pp);
        pa.u[1] = *reinterpret_cast<const uint4*>(pp + 16);
#pragma unroll
        for (int j = 0; j < 4; ++j) {
            FragB16 vf;
            const bf16_t* vp = vtp + (j * 16 + lrow) * LKP + ks * 32 + khalf;
            vf.u[0] = *reinterpret_cast<const uint4*>(vp);
            vf.u[1] = *reinterpret_cast<const uint4*>(vp + 16);
            oacc[j] = __builtin_amdgcn_wmma_f32_16x16x32_bf16(
                false, pa.v, false, vf.v, (short)0, oacc[j], false, false);
        }
    }

    // --- store O as bf16 into [B, L, C] for the output projection GEMM ---
#pragma unroll
    for (int j = 0; j < 4; ++j) {
        int col = h * HDIM + j * 16 + lrow;
#pragma unroll
        for (int r = 0; r < 8; ++r) {
            int gr = rrow0 + r;
            if (gr < SEQ_L)
                Ob[((size_t)(b * SEQ_L) + gr) * CDIM + col] = (bf16_t)oacc[j][r];
        }
    }
}

// ---------------------------------------------------------------------------
// Host orchestration
// ---------------------------------------------------------------------------
extern "C" void kernel_launch(void* const* d_in, const int* in_sizes, int n_in,
                              void* d_out, int out_size, void* d_ws, size_t ws_size,
                              hipStream_t stream) {
    (void)in_sizes; (void)n_in; (void)out_size; (void)ws_size;
    const float* x         = (const float*)d_in[0];
    const float* y         = (const float*)d_in[1];
    const float* freqs     = (const float*)d_in[2];
    const float* attn_bias = (const float*)d_in[3];
    const float* Wq        = (const float*)d_in[4];
    const float* q_bias    = (const float*)d_in[5];
    const float* Wkv       = (const float*)d_in[6];
    const float* v_bias    = (const float*)d_in[7];
    const float* Wproj     = (const float*)d_in[8];
    const float* b_proj    = (const float*)d_in[9];
    const float* scale_mul = (const float*)d_in[10];
    float* out = (float*)d_out;

    const int B = BATCH, L = SEQ_L, Lk = SEQ_LK, C = CDIM, H = HEADS, D = HDIM;
    const size_t nX   = (size_t)B * L * C;        // 13,959,168
    const size_t nY   = (size_t)B * Lk * C;
    const size_t nWq  = (size_t)C * C;
    const size_t nWkv = (size_t)2 * C * C;

    char* ws = (char*)d_ws;
    size_t off = 0;
    auto alloc = [&](size_t bytes) -> void* {
        void* p = ws + off;
        off = (off + bytes + 255) & ~(size_t)255;
        return p;
    };
    bf16_t* xb   = (bf16_t*)alloc(nX * 2);
    bf16_t* yb   = (bf16_t*)alloc(nY * 2);
    bf16_t* wqb  = (bf16_t*)alloc(nWq * 2);
    bf16_t* wkvb = (bf16_t*)alloc(nWkv * 2);
    bf16_t* wpb  = (bf16_t*)alloc(nWq * 2);
    float*  Qf   = (float*)alloc(nX * 4);
    float*  KVf  = (float*)alloc((size_t)B * Lk * 2 * C * 4);
    bf16_t* Qb   = (bf16_t*)alloc(nX * 2);
    bf16_t* Kb   = (bf16_t*)alloc((size_t)B * H * Lk * D * 2);
    size_t  vtBytes = (size_t)B * H * D * LKP * 2;
    bf16_t* Vt   = (bf16_t*)alloc(vtBytes);
    bf16_t* Ob   = (bf16_t*)alloc(nX * 2);

    auto cvt = [&](const float* src, bf16_t* dst, size_t n) {
        int blocks = (int)((n + 1023) / 1024);
        if (blocks > 4096) blocks = 4096;
        cvt_f32_bf16<<<blocks, 256, 0, stream>>>(src, dst, (int)n);
    };
    cvt(x, xb, nX);
    cvt(y, yb, nY);
    cvt(Wq, wqb, nWq);
    cvt(Wkv, wkvb, nWkv);
    cvt(Wproj, wpb, nWq);

    // Q projection: [B*L, C] = xb * Wq^T   (bias applied in epilogue)
    gemm_bf16_nt<<<dim3((B * L + 127) / 128, C / 64), 256, 0, stream>>>(
        xb, wqb, nullptr, Qf, B * L, C, C);
    // KV projection: [B*Lk, 2C] = yb * Wkv^T
    gemm_bf16_nt<<<dim3((B * Lk + 127) / 128, (2 * C) / 64), 256, 0, stream>>>(
        yb, wkvb, nullptr, KVf, B * Lk, 2 * C, C);

    // zero Vt (for Lk->LKP padding)
    (void)hipMemsetAsync(Vt, 0, vtBytes, stream);

    // epilogues: bias + norm + RoPE, bf16 repack
    q_epilogue<<<(B * L * H + 7) / 8, 256, 0, stream>>>(
        Qf, q_bias, freqs, scale_mul, Qb);
    kv_epilogue<<<(B * Lk * H + 7) / 8, 256, 0, stream>>>(
        KVf, v_bias, Kb, Vt);

    // fused attention
    attn_kernel<<<dim3((L + 127) / 128, H, B), 256, 0, stream>>>(
        Qb, Kb, Vt, attn_bias, Ob);

    // output projection with bias -> fp32 out
    gemm_bf16_nt<<<dim3((B * L + 127) / 128, C / 64), 256, 0, stream>>>(
        Ob, wpb, b_proj, out, B * L, C, C);
}
